// Dynamic_Pointing_Decoder_52673478918640
// MI455X (gfx1250) — matrix operand order repaired and checked by reading, not compiled, not run on hardware
//
#include <hip/hip_runtime.h>
#include <hip/hip_bf16.h>

#define BATCH 32
#define MLEN  2048
#define EMB   200
#define UDIM  400
#define NP    3200      // EMB*POOL
#define ITERS 4
#define KU    416       // UDIM padded to 13*32
#define K2    224       // EMB padded to 7*32
#define NT1   200       // NP/16 (one pool group per 16-wide N tile)
#define TM    32        // M rows per block (2 x 16-row WMMA subtiles)

typedef __attribute__((ext_vector_type(16))) __bf16 v16bf;
typedef __attribute__((ext_vector_type(8)))  float  v8f;
typedef __attribute__((ext_vector_type(4)))  unsigned int v4u;
typedef __attribute__((ext_vector_type(8)))  int v8i;
typedef __attribute__((ext_vector_type(4)))  int v4i;

struct Frag32B { uint4 lo, hi; };

__device__ __forceinline__ v16bf make_frag(uint4 lo, uint4 hi) {
    Frag32B t{lo, hi};
    return __builtin_bit_cast(v16bf, t);
}

__device__ __forceinline__ unsigned short f2bf(float x) {
    unsigned int u = __float_as_uint(x);
    unsigned int r = u + 0x7FFFu + ((u >> 16) & 1u);   // round to nearest even
    return (unsigned short)(r >> 16);
}

__device__ __forceinline__ float bf2f(unsigned short v) {
    return __uint_as_float((unsigned)v << 16);
}

__device__ __forceinline__ float sigmf(float x) { return 1.f / (1.f + expf(-x)); }

// ---------------------------------------------------------------------------
// Prep: convert fp32 matrix rows to bf16 with zero-padded K
// ---------------------------------------------------------------------------
__global__ void convert_pad(const float* __restrict__ src, unsigned short* __restrict__ dst,
                            long rows, int ksrc, int kdst, int kcopy) {
    long i = (long)blockIdx.x * blockDim.x + threadIdx.x;
    long total = rows * kdst;
    if (i >= total) return;
    long r = i / kdst;
    int  k = (int)(i % kdst);
    float v = (k < kcopy) ? src[r * ksrc + k] : 0.f;
    dst[i] = f2bf(v);
}

__global__ void init_state(const float* __restrict__ U, float* __restrict__ u0,
                           float* __restrict__ h, float* __restrict__ c) {
    int i = blockIdx.x * blockDim.x + threadIdx.x;
    if (i >= BATCH * UDIM) return;
    int b = i / UDIM, k = i % UDIM;
    u0[i] = U[(long)b * MLEN * UDIM + k];     // U[:,0,:]
    if (k < EMB) { h[b * EMB + k] = 0.f; c[b * EMB + k] = 0.f; }
}

// ---------------------------------------------------------------------------
// LSTM (tiny: B=32)
// ---------------------------------------------------------------------------
__global__ void lstm_gates(const float* __restrict__ us, const float* __restrict__ ue,
                           const float* __restrict__ h,
                           const float* __restrict__ Wih, const float* __restrict__ Whh,
                           const float* __restrict__ bih, const float* __restrict__ bhh,
                           float* __restrict__ g) {
    int i = blockIdx.x * blockDim.x + threadIdx.x;
    if (i >= BATCH * 4 * EMB) return;
    int b = i / (4 * EMB), j = i % (4 * EMB);
    const float* wi = Wih + (long)j * (2 * UDIM);
    const float* wh = Whh + (long)j * EMB;
    float acc = bih[j] + bhh[j];
    const float* x1 = us + b * UDIM;
    const float* x2 = ue + b * UDIM;
    for (int k = 0; k < UDIM; ++k) acc += x1[k] * wi[k];
    for (int k = 0; k < UDIM; ++k) acc += x2[k] * wi[UDIM + k];
    const float* hp = h + b * EMB;
    for (int k = 0; k < EMB; ++k) acc += hp[k] * wh[k];
    g[i] = acc;
}

__global__ void lstm_update(const float* __restrict__ g, float* __restrict__ h,
                            float* __restrict__ c) {
    int i = blockIdx.x * blockDim.x + threadIdx.x;
    if (i >= BATCH * EMB) return;
    int b = i / EMB, d = i % EMB;
    const float* gb = g + b * 4 * EMB;
    float ig = sigmf(gb[d]);
    float fg = sigmf(gb[EMB + d]);
    float gg = tanhf(gb[2 * EMB + d]);
    float og = sigmf(gb[3 * EMB + d]);
    float cn = fg * c[i] + ig * gg;
    c[i] = cn;
    h[i] = og * tanhf(cn);
}

// r = tanh([h, u_s, u_e] @ W1^T), per side
__global__ void compute_r(const float* __restrict__ h, const float* __restrict__ us,
                          const float* __restrict__ ue,
                          const float* __restrict__ W1s, const float* __restrict__ W1e,
                          float* __restrict__ rs, float* __restrict__ re) {
    int i = blockIdx.x * blockDim.x + threadIdx.x;
    if (i >= 2 * BATCH * EMB) return;
    int side = i / (BATCH * EMB);
    int rem = i % (BATCH * EMB);
    int b = rem / EMB, d = rem % EMB;
    const float* w = (side ? W1e : W1s) + (long)d * (EMB + 2 * UDIM);
    float acc = 0.f;
    const float* hp = h + b * EMB;
    for (int k = 0; k < EMB; ++k) acc += hp[k] * w[k];
    const float* u1 = us + b * UDIM;
    for (int k = 0; k < UDIM; ++k) acc += u1[k] * w[EMB + k];
    const float* u2 = ue + b * UDIM;
    for (int k = 0; k < UDIM; ++k) acc += u2[k] * w[EMB + UDIM + k];
    (side ? re : rs)[rem] = tanhf(acc);
}

// rc[b][n] = bm1[n] + sum_j r[b][j] * Wm1[n][400+j]   (folded r-part of m1)
__global__ void compute_rc(const float* __restrict__ rs, const float* __restrict__ re,
                           const float* __restrict__ Wm1s, const float* __restrict__ Wm1e,
                           const float* __restrict__ bm1s, const float* __restrict__ bm1e,
                           float* __restrict__ rcs, float* __restrict__ rce) {
    int i = blockIdx.x * blockDim.x + threadIdx.x;
    if (i >= 2 * BATCH * NP) return;
    int side = i / (BATCH * NP);
    int rem = i % (BATCH * NP);
    int b = rem / NP, n = rem % NP;
    const float* r = (side ? re : rs) + b * EMB;
    const float* w = (side ? Wm1e : Wm1s) + (long)n * 600 + UDIM;
    float acc = (side ? bm1e : bm1s)[n];
    for (int j = 0; j < EMB; ++j) acc += r[j] * w[j];
    (side ? rce : rcs)[rem] = acc;
}

// ---------------------------------------------------------------------------
// Fused HMN: TDM-staged A tile -> m1 (WMMA bf16) -> maxpool -> m2 (WMMA bf16)
//            -> maxpool -> m3 -> masked score.
// One block = one (batch, 32-row M tile). 256 threads = 8 waves; each wave
// holds TWO 16-row A-fragment sets so every weight B-tile load feeds 2 WMMAs.
// ---------------------------------------------------------------------------
__global__ __launch_bounds__(256) void hmn_fused(
    const unsigned short* __restrict__ Ub,    // [B*M][KU] bf16 (padded)
    const unsigned short* __restrict__ W1b,   // [NP][KU] bf16 (U-part of Wm1)
    const float* __restrict__ rc,             // [B][NP]  (r-part + bm1)
    const unsigned short* __restrict__ W2b,   // [NP][K2] bf16
    const float* __restrict__ bm2,            // [NP]
    const float* __restrict__ Wm3,            // [16][400]
    const float* __restrict__ bm3,            // [16]
    const int* __restrict__ doclen,           // [B]
    float* __restrict__ out,                  // d_out
    int iter, int side)
{
    __shared__ __align__(16) unsigned short Alds[TM * KU];   // 26624 B
    __shared__ __align__(16) unsigned short m1b[TM * K2];    // 14336 B
    __shared__ __align__(16) unsigned short m2b[TM * K2];    // 14336 B

    const int mt = blockIdx.x, b = blockIdx.y;
    const int t = threadIdx.x, wave = t >> 5, lane = t & 31;
    const long rowbase = (long)b * MLEN + (long)mt * TM;

    // ---- Stage the contiguous 32x832B A-tile via the Tensor Data Mover ----
    if (wave == 0) {
        unsigned long long ga = (unsigned long long)(uintptr_t)(Ub + rowbase * KU);
        unsigned la = (unsigned)(uintptr_t)(void*)Alds;      // LDS byte offset (flat addr low bits)
        const unsigned TB = TM * KU * 2;                     // 26624 bytes (1D tile)
        // D# group0: count=1 | lds_addr | global_addr[56:0] | type=2
        v4u g0 = { 1u, la, (unsigned)ga,
                   (unsigned)((ga >> 32) & 0x1FFFFFFull) | (2u << 30) };
        // D# group1: mask=0,data_size=1B | dim0=TB | dim1=1 | tile0=TB | stride0=TB
        v8i g1 = { 0,
                   (int)(TB << 16),                   // tensor_dim0[15:0]  @ [31:16]
                   (int)((TB >> 16) | 0x10000u),      // dim0[31:16] | tensor_dim1=1
                   (int)(TB << 16),                   // dim1[31:16]=0 | tile_dim0=TB
                   0,                                 // tile_dim1=0, tile_dim2=0
                   (int)TB,                           // tensor_dim0_stride[31:0]
                   0, 0 };
        v4i gz = { 0, 0, 0, 0 };
#if defined(__clang_major__) && (__clang_major__ >= 23)
        v8i z8 = { 0, 0, 0, 0, 0, 0, 0, 0 };
        __builtin_amdgcn_tensor_load_to_lds(g0, g1, gz, gz, z8, 0);
#else
        __builtin_amdgcn_tensor_load_to_lds(g0, g1, gz, gz, 0);
#endif
        __builtin_amdgcn_s_wait_tensorcnt(0);
    }
    // Zero the bf16 padding columns of m1b (A-operand padding for the m2 WMMA)
    for (int i = t; i < TM * (K2 - EMB); i += 256) {
        int r = i / (K2 - EMB), k = i % (K2 - EMB);
        m1b[r * K2 + EMB + k] = 0;
    }
    __syncthreads();

    const int arow = lane & 15;                 // A row within 16-row subtile
    const int gh   = (lane >= 16) ? 8 : 0;      // high-lane K offset (A) / row offset (C)
    const int bcol = lane & 15;                 // B column (N)
    const int kb   = (lane >= 16) ? 8 : 0;      // B K base

    // ---- m1: two A-fragment sets (rows 0-15, 16-31), built once ----
    v16bf a1[2][13];
#pragma unroll
    for (int s = 0; s < 2; ++s)
#pragma unroll
        for (int kt = 0; kt < 13; ++kt) {
            const unsigned short* p = &Alds[(s * 16 + arow) * KU + kt * 32 + gh];
            a1[s][kt] = make_frag(*(const uint4*)p, *(const uint4*)(p + 16));
        }

    for (int nt = wave; nt < NT1; nt += 8) {
        v8f acc0 = {}, acc1 = {};
        const unsigned short* wrow = W1b + (long)(nt * 16 + bcol) * KU + kb;
#pragma unroll
        for (int kt = 0; kt < 13; ++kt) {
            v16bf bfr = make_frag(*(const uint4*)(wrow + kt * 32),
                                  *(const uint4*)(wrow + kt * 32 + 16));
            acc0 = __builtin_amdgcn_wmma_f32_16x16x32_bf16(
                false, a1[0][kt], false, bfr, (short)0, acc0, false, false);
            acc1 = __builtin_amdgcn_wmma_f32_16x16x32_bf16(
                false, a1[1][kt], false, bfr, (short)0, acc1, false, false);
        }
        float rcv = rc[b * NP + nt * 16 + bcol];
#pragma unroll
        for (int v = 0; v < 8; ++v) {
            float y0 = acc0[v] + rcv;           // subtile 0: row = v+gh
            float y1 = acc1[v] + rcv;           // subtile 1: row = 16+v+gh
            y0 = fmaxf(y0, __shfl_xor(y0, 1));
            y0 = fmaxf(y0, __shfl_xor(y0, 2));
            y0 = fmaxf(y0, __shfl_xor(y0, 4));
            y0 = fmaxf(y0, __shfl_xor(y0, 8));  // max over 16-wide pool group
            y1 = fmaxf(y1, __shfl_xor(y1, 1));
            y1 = fmaxf(y1, __shfl_xor(y1, 2));
            y1 = fmaxf(y1, __shfl_xor(y1, 4));
            y1 = fmaxf(y1, __shfl_xor(y1, 8));
            if (bcol == 0) {
                m1b[(v + gh) * K2 + nt]      = f2bf(y0);
                m1b[(16 + v + gh) * K2 + nt] = f2bf(y1);
            }
        }
    }
    __syncthreads();

    // ---- m2: A = m1 (2 x 16x224 bf16) ----
    v16bf a2[2][7];
#pragma unroll
    for (int s = 0; s < 2; ++s)
#pragma unroll
        for (int kt = 0; kt < 7; ++kt) {
            const unsigned short* p = &m1b[(s * 16 + arow) * K2 + kt * 32 + gh];
            a2[s][kt] = make_frag(*(const uint4*)p, *(const uint4*)(p + 16));
        }

    for (int nt = wave; nt < NT1; nt += 8) {
        v8f acc0 = {}, acc1 = {};
        const unsigned short* wrow = W2b + (long)(nt * 16 + bcol) * K2 + kb;
#pragma unroll
        for (int kt = 0; kt < 7; ++kt) {
            v16bf bfr = make_frag(*(const uint4*)(wrow + kt * 32),
                                  *(const uint4*)(wrow + kt * 32 + 16));
            acc0 = __builtin_amdgcn_wmma_f32_16x16x32_bf16(
                false, a2[0][kt], false, bfr, (short)0, acc0, false, false);
            acc1 = __builtin_amdgcn_wmma_f32_16x16x32_bf16(
                false, a2[1][kt], false, bfr, (short)0, acc1, false, false);
        }
        float bv = bm2[nt * 16 + bcol];
#pragma unroll
        for (int v = 0; v < 8; ++v) {
            float y0 = acc0[v] + bv;
            float y1 = acc1[v] + bv;
            y0 = fmaxf(y0, __shfl_xor(y0, 1));
            y0 = fmaxf(y0, __shfl_xor(y0, 2));
            y0 = fmaxf(y0, __shfl_xor(y0, 4));
            y0 = fmaxf(y0, __shfl_xor(y0, 8));
            y1 = fmaxf(y1, __shfl_xor(y1, 1));
            y1 = fmaxf(y1, __shfl_xor(y1, 2));
            y1 = fmaxf(y1, __shfl_xor(y1, 4));
            y1 = fmaxf(y1, __shfl_xor(y1, 8));
            if (bcol == 0) {
                m2b[(v + gh) * K2 + nt]      = f2bf(y0);
                m2b[(16 + v + gh) * K2 + nt] = f2bf(y1);
            }
        }
    }
    __syncthreads();

    // ---- m3: per (row, p) dot of length 400 over [m1, m2], maxout over p ----
#pragma unroll
    for (int rr = 0; rr < 2; ++rr) {
        int row = (t >> 4) + 16 * rr;
        int p = t & 15;
        const float* w = Wm3 + p * (2 * EMB);
        float acc = bm3[p];
        for (int k = 0; k < EMB; ++k) acc += bf2f(m1b[row * K2 + k]) * w[k];
        for (int k = 0; k < EMB; ++k) acc += bf2f(m2b[row * K2 + k]) * w[EMB + k];
        acc = fmaxf(acc, __shfl_xor(acc, 1));
        acc = fmaxf(acc, __shfl_xor(acc, 2));
        acc = fmaxf(acc, __shfl_xor(acc, 4));
        acc = fmaxf(acc, __shfl_xor(acc, 8));
        if (p == 0) {
            int m = mt * TM + row;
            float sc = (m < doclen[b]) ? acc : -1000000000.0f;
            out[64 + ((long)side * BATCH * ITERS + (long)b * ITERS + iter) * MLEN + m] = sc;
        }
    }
}

// ---------------------------------------------------------------------------
// Per-(batch, side) argmax over masked scores + gather U[b, idx] for next iters
// ---------------------------------------------------------------------------
__global__ void argmax_gather(const float* __restrict__ dout, const float* __restrict__ U,
                              float* __restrict__ us_next, float* __restrict__ ue_next,
                              float* __restrict__ out_se, int iter) {
    int b = blockIdx.x, side = blockIdx.y;
    const float* sc = dout + 64 + ((long)side * BATCH * ITERS + (long)b * ITERS + iter) * MLEN;
    __shared__ float bv[256];
    __shared__ int   bi[256];
    int t = threadIdx.x;
    float best = -3.4e38f;
    int besti = 0;
    for (int m = t; m < MLEN; m += 256) {
        float v = sc[m];
        if (v > best) { best = v; besti = m; }   // strided scan keeps lowest index on ties
    }
    bv[t] = best; bi[t] = besti;
    __syncthreads();
    for (int s = 128; s > 0; s >>= 1) {
        if (t < s) {
            if (bv[t + s] > bv[t] || (bv[t + s] == bv[t] && bi[t + s] < bi[t])) {
                bv[t] = bv[t + s]; bi[t] = bi[t + s];
            }
        }
        __syncthreads();
    }
    int idx = bi[0];
    float* dst = (side ? ue_next : us_next) + b * UDIM;
    const float* srow = U + ((long)b * MLEN + idx) * UDIM;
    for (int k = t; k < UDIM; k += 256) dst[k] = srow[k];
    if (t == 0 && out_se) out_se[side * BATCH + b] = (float)idx;
}

// ---------------------------------------------------------------------------
extern "C" void kernel_launch(void* const* d_in, const int* in_sizes, int n_in,
                              void* d_out, int out_size, void* d_ws, size_t ws_size,
                              hipStream_t stream) {
    (void)in_sizes; (void)n_in; (void)out_size; (void)ws_size;
    const float* U   = (const float*)d_in[0];
    const int*   dl  = (const int*)d_in[1];
    const float* Wih = (const float*)d_in[2];
    const float* Whh = (const float*)d_in[3];
    const float* bih = (const float*)d_in[4];
    const float* bhh = (const float*)d_in[5];
    const float* W1s  = (const float*)d_in[6];
    const float* Wm1s = (const float*)d_in[7];
    const float* bm1s = (const float*)d_in[8];
    const float* Wm2s = (const float*)d_in[9];
    const float* bm2s = (const float*)d_in[10];
    const float* Wm3s = (const float*)d_in[11];
    const float* bm3s = (const float*)d_in[12];
    const float* W1e  = (const float*)d_in[13];
    const float* Wm1e = (const float*)d_in[14];
    const float* bm1e = (const float*)d_in[15];
    const float* Wm2e = (const float*)d_in[16];
    const float* bm2e = (const float*)d_in[17];
    const float* Wm3e = (const float*)d_in[18];
    const float* bm3e = (const float*)d_in[19];
    float* out = (float*)d_out;

    // Workspace bump allocator (256B aligned)
    char* wp = (char*)d_ws;
    auto alloc = [&](size_t bytes) -> void* {
        void* r = (void*)wp;
        wp += (bytes + 255) & ~(size_t)255;
        return r;
    };
    unsigned short* Ub     = (unsigned short*)alloc((size_t)BATCH * MLEN * KU * 2);
    unsigned short* W1bS   = (unsigned short*)alloc((size_t)NP * KU * 2);
    unsigned short* W1bE   = (unsigned short*)alloc((size_t)NP * KU * 2);
    unsigned short* W2bS   = (unsigned short*)alloc((size_t)NP * K2 * 2);
    unsigned short* W2bE   = (unsigned short*)alloc((size_t)NP * K2 * 2);
    float* h   = (float*)alloc(BATCH * EMB * 4);
    float* c   = (float*)alloc(BATCH * EMB * 4);
    float* g   = (float*)alloc(BATCH * 4 * EMB * 4);
    float* u0  = (float*)alloc(BATCH * UDIM * 4);
    float* usg1 = (float*)alloc(BATCH * UDIM * 4);
    float* usg2 = (float*)alloc(BATCH * UDIM * 4);
    float* ueg1 = (float*)alloc(BATCH * UDIM * 4);
    float* ueg2 = (float*)alloc(BATCH * UDIM * 4);
    float* rs  = (float*)alloc(BATCH * EMB * 4);
    float* re  = (float*)alloc(BATCH * EMB * 4);
    float* rcs = (float*)alloc(BATCH * NP * 4);
    float* rce = (float*)alloc(BATCH * NP * 4);

    const int TB = 256;
    auto nb = [](long total, int tb) { return (unsigned)((total + tb - 1) / tb); };

    // One-time prep: bf16 conversions (U fits L2; subsequent passes hit cache)
    convert_pad<<<nb((long)BATCH * MLEN * KU, TB), TB, 0, stream>>>(U, Ub, (long)BATCH * MLEN, UDIM, KU, UDIM);
    convert_pad<<<nb((long)NP * KU, TB), TB, 0, stream>>>(Wm1s, W1bS, NP, 600, KU, UDIM);
    convert_pad<<<nb((long)NP * KU, TB), TB, 0, stream>>>(Wm1e, W1bE, NP, 600, KU, UDIM);
    convert_pad<<<nb((long)NP * K2, TB), TB, 0, stream>>>(Wm2s, W2bS, NP, EMB, K2, EMB);
    convert_pad<<<nb((long)NP * K2, TB), TB, 0, stream>>>(Wm2e, W2bE, NP, EMB, K2, EMB);
    init_state<<<nb(BATCH * UDIM, TB), TB, 0, stream>>>(U, u0, h, c);

    // Reference's one-iteration-delayed pointer rotation:
    float* us_im1 = u0; float* ue_im1 = u0;
    float* us_i   = u0; float* ue_i   = u0;

    for (int iter = 0; iter < ITERS; ++iter) {
        lstm_gates<<<nb(BATCH * 4 * EMB, TB), TB, 0, stream>>>(us_im1, ue_im1, h, Wih, Whh, bih, bhh, g);
        lstm_update<<<nb(BATCH * EMB, TB), TB, 0, stream>>>(g, h, c);
        compute_r<<<nb(2 * BATCH * EMB, TB), TB, 0, stream>>>(h, us_im1, ue_im1, W1s, W1e, rs, re);
        compute_rc<<<nb((long)2 * BATCH * NP, TB), TB, 0, stream>>>(rs, re, Wm1s, Wm1e, bm1s, bm1e, rcs, rce);

        hmn_fused<<<dim3(MLEN / TM, BATCH), 256, 0, stream>>>(
            Ub, W1bS, rcs, W2bS, bm2s, Wm3s, bm3s, dl, out, iter, 0);
        hmn_fused<<<dim3(MLEN / TM, BATCH), 256, 0, stream>>>(
            Ub, W1bE, rce, W2bE, bm2e, Wm3e, bm3e, dl, out, iter, 1);

        float* us_tgt = (iter & 1) ? usg2 : usg1;
        float* ue_tgt = (iter & 1) ? ueg2 : ueg1;
        argmax_gather<<<dim3(BATCH, 2), 256, 0, stream>>>(
            out, U, us_tgt, ue_tgt, (iter == ITERS - 1) ? out : nullptr, iter);

        us_im1 = us_i; ue_im1 = ue_i;
        us_i = us_tgt; ue_i = ue_tgt;
    }
}
